// TransformerBlock_86320252715595
// MI455X (gfx1250) — compile-verified, matrix operand order checked by reading, not run on hardware
//
#include <hip/hip_runtime.h>
#include <hip/hip_bf16.h>

// ---------------------------------------------------------------------------
// Transformer block forward for MI455X (gfx1250), wave32 + WMMA bf16.
// Matmuls: v_wmma_f32_16x16x32_bf16 (fp32 accumulate).
// LDS staging: double-buffered global_load_async_to_lds_b128 (ASYNCcnt)
// pipeline so the tensor units overlap with the DMA.
// ---------------------------------------------------------------------------

typedef __attribute__((ext_vector_type(16))) __bf16 v16bf;
typedef __attribute__((ext_vector_type(8)))  float  v8f;

union ABFrag { v16bf v; __bf16 e[16]; uint4 q[2]; };
union AccFrag { v8f v; float e[8]; };
union BF16x4 { __bf16 h[4]; uint2 u; };

#define TB_B    2
#define TB_S    2048
#define TB_D    1024
#define TB_H    16
#define TB_DK   64
#define TB_DFF  4096

// --- CDNA5 async copy: global -> LDS, 16 bytes per lane, tracked by ASYNCcnt.
template <typename T>
__device__ __forceinline__ unsigned lds_addr(T* p) {
    return (unsigned)(unsigned long long)(__attribute__((address_space(3))) T*)p;
}
__device__ __forceinline__ void async_copy_b128(unsigned lds_off, const void* gaddr) {
    asm volatile("global_load_async_to_lds_b128 %0, %1, off"
                 :: "v"(lds_off), "v"(gaddr) : "memory");
}
__device__ __forceinline__ void async_wait0() {
    asm volatile("s_wait_asynccnt 0x0" ::: "memory");
}

// ---------------------------------------------------------------------------
// fp32 -> bf16 conversion (vectorized: float4 in, 4x bf16 out)
// ---------------------------------------------------------------------------
__global__ __launch_bounds__(256) void cvt_f32_bf16(const float* __restrict__ in,
                                                    __bf16* __restrict__ out, int n4) {
    int i = blockIdx.x * 256 + threadIdx.x;
    if (i < n4) {
        float4 v = ((const float4*)in)[i];
        BF16x4 pk;
        pk.h[0] = (__bf16)v.x; pk.h[1] = (__bf16)v.y;
        pk.h[2] = (__bf16)v.z; pk.h[3] = (__bf16)v.w;
        ((uint2*)out)[i] = pk.u;
    }
}

// ---------------------------------------------------------------------------
// Generic bf16 GEMM: C[M,N] = A[M,K] @ B[K,N], fp32 accum, templated epilogue.
// Block tile 128(M) x 128(N), BK=32. 256 threads = 8 waves in a 4(M)x2(N)
// grid; each wave runs 2 A-frags x 4 B-frags = 8 WMMAs per K-step.
// Double-buffered async global->LDS staging (tile k+1 overlaps compute of k).
//   EPI==0 : QKV  -> bf16 store permuted to [B,H,S,dk]      (outB)
//   EPI==1 : f32  -> acc + bias[n] + res[m*N+n]             (outF)
//   EPI==2 : bf16 -> gelu(acc + bias[n])  (exact, erff)     (outB)
// ---------------------------------------------------------------------------
template <int EPI>
__global__ __launch_bounds__(256)
void gemm_bf16(const __bf16* __restrict__ A, const __bf16* __restrict__ Bm,
               int M, int N, int K,
               const float* __restrict__ bias, const float* __restrict__ res,
               float* __restrict__ outF, __bf16* __restrict__ outB) {
    __shared__ __bf16 As[2][128][32];   // 2 x 8 KB
    __shared__ __bf16 Bs[2][32][128];   // 2 x 8 KB

    const int tid  = threadIdx.x;
    const int lane = tid & 31;
    const int wave = tid >> 5;
    const int wm   = wave & 3;   // 0..3  (M, 32 rows)
    const int wn   = wave >> 2;  // 0..1  (N, 64 cols)
    const int m0   = blockIdx.x * 128;
    const int n0   = blockIdx.y * 128;

    AccFrag acc[2][4] = {};
    const int koff = (lane >> 4) * 8;   // 0 or 8

    auto stage = [&](int buf, int k0) {
#pragma unroll
        for (int c = 0; c < 2; ++c) {
            const int idx  = tid + c * 256;     // 0..511
            const int arow = idx >> 2;          // 0..127
            const int acol = (idx & 3) * 8;     // 0..24
            async_copy_b128(lds_addr(&As[buf][arow][acol]),
                            &A[(size_t)(m0 + arow) * K + k0 + acol]);
            const int brow = idx >> 4;          // 0..31
            const int bcol = (idx & 15) * 8;    // 0..120
            async_copy_b128(lds_addr(&Bs[buf][brow][bcol]),
                            &Bm[(size_t)(k0 + brow) * N + n0 + bcol]);
        }
    };

    const int nk = K >> 5;
    stage(0, 0);
    for (int ki = 0; ki < nk; ++ki) {
        const int buf = ki & 1;
        async_wait0();        // tile ki resident in LDS
        __syncthreads();      // all waves done with buf^1 reads + ki stores seen
        if (ki + 1 < nk) stage(buf ^ 1, (ki + 1) << 5);

        // ---- A fragments (16x32 bf16, ISA layout)
        ABFrag a[2];
#pragma unroll
        for (int mt = 0; mt < 2; ++mt) {
            const int r = wm * 32 + mt * 16 + (lane & 15);
            a[mt].q[0] = *(const uint4*)&As[buf][r][koff];
            a[mt].q[1] = *(const uint4*)&As[buf][r][16 + koff];
        }
#pragma unroll
        for (int nt = 0; nt < 4; ++nt) {
            ABFrag b;
            const int c = wn * 64 + nt * 16 + (lane & 15);
#pragma unroll
            for (int j = 0; j < 8; ++j) {
                b.e[j]     = Bs[buf][koff + j][c];
                b.e[8 + j] = Bs[buf][16 + koff + j][c];
            }
#pragma unroll
            for (int mt = 0; mt < 2; ++mt) {
                acc[mt][nt].v = __builtin_amdgcn_wmma_f32_16x16x32_bf16(
                    false, a[mt].v, false, b.v, (short)0, acc[mt][nt].v, false, false);
            }
        }
    }

    // ---- epilogue: lane L holds N=L&15, VGPR i holds M = i + 8*(L>=16)
    const int rof = (lane >> 4) * 8;
    const int cl  = lane & 15;
#pragma unroll
    for (int mt = 0; mt < 2; ++mt) {
#pragma unroll
        for (int nt = 0; nt < 4; ++nt) {
            const int n = n0 + wn * 64 + nt * 16 + cl;
#pragma unroll
            for (int i = 0; i < 8; ++i) {
                const int m = m0 + wm * 32 + mt * 16 + rof + i;
                float v = acc[mt][nt].e[i];
                if (EPI == 0) {
                    // permute to [B, H, S, dk]
                    const int bb = m / TB_S, s = m % TB_S;
                    const int h = n >> 6, d = n & 63;
                    outB[((((size_t)bb * TB_H + h) * TB_S + s) << 6) + d] = (__bf16)v;
                } else if (EPI == 1) {
                    outF[(size_t)m * N + n] = v + bias[n] + res[(size_t)m * N + n];
                } else {
                    float t = v + bias[n];
                    float g = 0.5f * t * (1.0f + erff(t * 0.70710678118654752f));
                    outB[(size_t)m * N + n] = (__bf16)g;
                }
            }
        }
    }
}

// ---------------------------------------------------------------------------
// scores = (Q @ K^T) * scale, per (b,h).  Q,K: [BH, S, 64] bf16.
// Both operands are contiguous along d_k -> direct 16B register loads.
// grid: (S/16, S/128, B*H), block 256.
// ---------------------------------------------------------------------------
__global__ __launch_bounds__(256)
void attn_scores(const __bf16* __restrict__ Qh, const __bf16* __restrict__ Kh,
                 float* __restrict__ attn, float scale) {
    const int lane = threadIdx.x & 31;
    const int wave = threadIdx.x >> 5;
    const int bh   = blockIdx.z;
    const __bf16* Q = Qh + (size_t)bh * TB_S * TB_DK;
    const __bf16* Kk = Kh + (size_t)bh * TB_S * TB_DK;
    const int q0 = blockIdx.x * 16;
    const int n0 = blockIdx.y * 128 + wave * 16;

    AccFrag acc = {};
    const int r    = lane & 15;
    const int koff = (lane >> 4) * 8;
#pragma unroll
    for (int k0 = 0; k0 < TB_DK; k0 += 32) {
        ABFrag a, b;
        a.q[0] = *(const uint4*)&Q[(size_t)(q0 + r) * TB_DK + k0 + koff];
        a.q[1] = *(const uint4*)&Q[(size_t)(q0 + r) * TB_DK + k0 + 16 + koff];
        // B = K^T : column n of B is row n of K, contiguous in k
        b.q[0] = *(const uint4*)&Kk[(size_t)(n0 + r) * TB_DK + k0 + koff];
        b.q[1] = *(const uint4*)&Kk[(size_t)(n0 + r) * TB_DK + k0 + 16 + koff];
        acc.v = __builtin_amdgcn_wmma_f32_16x16x32_bf16(
            false, a.v, false, b.v, (short)0, acc.v, false, false);
    }
    float* out = attn + (size_t)bh * TB_S * TB_S;
    const int rof = (lane >> 4) * 8;
#pragma unroll
    for (int i = 0; i < 8; ++i)
        out[(size_t)(q0 + rof + i) * TB_S + n0 + (lane & 15)] = acc.e[i] * scale;
}

// ---------------------------------------------------------------------------
// Row softmax in place, register-resident: read once (b128), write once.
// One block per row of length S=2048: 256 threads x 8 floats.
// ---------------------------------------------------------------------------
__global__ __launch_bounds__(256)
void softmax_rows(float* __restrict__ attn) {
    __shared__ float red[256];
    float4* p4 = (float4*)(attn + (size_t)blockIdx.x * TB_S);
    const int tid = threadIdx.x;

    float4 a = p4[tid];
    float4 b = p4[tid + 256];

    float mx = fmaxf(fmaxf(fmaxf(a.x, a.y), fmaxf(a.z, a.w)),
                     fmaxf(fmaxf(b.x, b.y), fmaxf(b.z, b.w)));
    red[tid] = mx;  __syncthreads();
    for (int o = 128; o > 0; o >>= 1) { if (tid < o) red[tid] = fmaxf(red[tid], red[tid + o]); __syncthreads(); }
    mx = red[0];  __syncthreads();

    a.x = __expf(a.x - mx); a.y = __expf(a.y - mx); a.z = __expf(a.z - mx); a.w = __expf(a.w - mx);
    b.x = __expf(b.x - mx); b.y = __expf(b.y - mx); b.z = __expf(b.z - mx); b.w = __expf(b.w - mx);
    float sum = (a.x + a.y) + (a.z + a.w) + (b.x + b.y) + (b.z + b.w);
    red[tid] = sum;  __syncthreads();
    for (int o = 128; o > 0; o >>= 1) { if (tid < o) red[tid] += red[tid + o]; __syncthreads(); }
    float inv = 1.0f / red[0];

    a.x *= inv; a.y *= inv; a.z *= inv; a.w *= inv;
    b.x *= inv; b.y *= inv; b.z *= inv; b.w *= inv;
    p4[tid] = a;
    p4[tid + 256] = b;
}

// ---------------------------------------------------------------------------
// context = attn @ V per (b,h); attn is f32 (converted to bf16 on the fly),
// V: [BH, S, 64] bf16 double-buffered async in LDS.
// grid: (S/64, B*H), block 256 = 8 waves: 2 groups of 32 rows x 4 d-tiles;
// each wave: 2 M-frags x 1 B-frag = 2 WMMAs per K-step.
// ---------------------------------------------------------------------------
__global__ __launch_bounds__(256)
void attn_context(const float* __restrict__ attn, const __bf16* __restrict__ Vh,
                  __bf16* __restrict__ ctx) {
    __shared__ __bf16 Vs[2][32][64];
    const int tid = threadIdx.x, lane = tid & 31, wave = tid >> 5;
    const int wn = wave & 3, wmw = wave >> 2;
    const int bh = blockIdx.y;
    const int bb = bh >> 4, h = bh & 15;
    const float* Ap = attn + (size_t)bh * TB_S * TB_S;
    const __bf16* V = Vh + (size_t)bh * TB_S * TB_DK;
    const int m0 = blockIdx.x * 64 + wmw * 32;
    const int n0 = wn * 16;

    AccFrag acc[2] = {};
    const int r    = lane & 15;
    const int koff = (lane >> 4) * 8;
    const int vr   = tid >> 3, vc = (tid & 7) * 8;

    // prologue: stage tile 0
    async_copy_b128(lds_addr(&Vs[0][vr][vc]), &V[(size_t)vr * TB_DK + vc]);

    const int nk = TB_S >> 5;
    for (int ki = 0; ki < nk; ++ki) {
        const int buf = ki & 1;
        const int k0  = ki << 5;
        async_wait0();
        __syncthreads();
        if (ki + 1 < nk)
            async_copy_b128(lds_addr(&Vs[buf ^ 1][vr][vc]),
                            &V[(size_t)(k0 + 32 + vr) * TB_DK + vc]);

        ABFrag b;
        const int c = n0 + r;
#pragma unroll
        for (int j = 0; j < 8; ++j) {
            b.e[j]     = Vs[buf][koff + j][c];
            b.e[8 + j] = Vs[buf][16 + koff + j][c];
        }
#pragma unroll
        for (int mt = 0; mt < 2; ++mt) {
            ABFrag a;
            const float* ar = Ap + (size_t)(m0 + mt * 16 + r) * TB_S + k0;
#pragma unroll
            for (int j = 0; j < 8; ++j) {
                a.e[j]     = (__bf16)ar[koff + j];
                a.e[8 + j] = (__bf16)ar[16 + koff + j];
            }
            acc[mt].v = __builtin_amdgcn_wmma_f32_16x16x32_bf16(
                false, a.v, false, b.v, (short)0, acc[mt].v, false, false);
        }
    }

    const int rof = (lane >> 4) * 8;
#pragma unroll
    for (int mt = 0; mt < 2; ++mt) {
#pragma unroll
        for (int i = 0; i < 8; ++i) {
            int s = m0 + mt * 16 + rof + i;
            int d = n0 + (lane & 15);
            ctx[((size_t)bb * TB_S + s) * TB_D + h * TB_DK + d] = (__bf16)acc[mt].e[i];
        }
    }
}

// ---------------------------------------------------------------------------
// LayerNorm over rows of length D=1024; register-resident float4 per thread.
// Writes f32 and (optionally) bf16.
// ---------------------------------------------------------------------------
__global__ __launch_bounds__(256)
void layernorm_rows(const float* __restrict__ in, const float* __restrict__ g,
                    const float* __restrict__ bta, float* __restrict__ outF,
                    __bf16* __restrict__ outB) {
    __shared__ float red[256];
    const size_t row = blockIdx.x;
    const int tid = threadIdx.x;

    float4 xv = ((const float4*)(in + row * (size_t)TB_D))[tid];

    float s = (xv.x + xv.y) + (xv.z + xv.w);
    red[tid] = s;  __syncthreads();
    for (int o = 128; o > 0; o >>= 1) { if (tid < o) red[tid] += red[tid + o]; __syncthreads(); }
    float mu = red[0] * (1.0f / TB_D);  __syncthreads();

    float dx = xv.x - mu, dy = xv.y - mu, dz = xv.z - mu, dw = xv.w - mu;
    red[tid] = dx * dx + dy * dy + dz * dz + dw * dw;  __syncthreads();
    for (int o = 128; o > 0; o >>= 1) { if (tid < o) red[tid] += red[tid + o]; __syncthreads(); }
    float rstd = rsqrtf(red[0] * (1.0f / TB_D) + 1e-5f);

    float4 gv = ((const float4*)g)[tid];
    float4 bv = ((const float4*)bta)[tid];
    float4 y;
    y.x = dx * rstd * gv.x + bv.x;
    y.y = dy * rstd * gv.y + bv.y;
    y.z = dz * rstd * gv.z + bv.z;
    y.w = dw * rstd * gv.w + bv.w;
    ((float4*)(outF + row * (size_t)TB_D))[tid] = y;
    if (outB) {
        BF16x4 pk;
        pk.h[0] = (__bf16)y.x; pk.h[1] = (__bf16)y.y;
        pk.h[2] = (__bf16)y.z; pk.h[3] = (__bf16)y.w;
        ((uint2*)(outB + row * (size_t)TB_D))[tid] = pk.u;
    }
}

// ---------------------------------------------------------------------------
extern "C" void kernel_launch(void* const* d_in, const int* in_sizes, int n_in,
                              void* d_out, int out_size, void* d_ws, size_t ws_size,
                              hipStream_t stream) {
    const float* x    = (const float*)d_in[0];
    const float* w_q  = (const float*)d_in[1];
    const float* w_k  = (const float*)d_in[2];
    const float* w_v  = (const float*)d_in[3];
    const float* w_o  = (const float*)d_in[4];
    const float* b_o  = (const float*)d_in[5];
    const float* w1   = (const float*)d_in[6];
    const float* b1   = (const float*)d_in[7];
    const float* w2   = (const float*)d_in[8];
    const float* b2   = (const float*)d_in[9];
    const float* ln1g = (const float*)d_in[10];
    const float* ln1b = (const float*)d_in[11];
    const float* ln2g = (const float*)d_in[12];
    const float* ln2b = (const float*)d_in[13];

    const size_t NT  = (size_t)TB_B * TB_S;       // 4096 tokens
    const size_t ND  = NT * TB_D;                 // 4,194,304
    const size_t NF  = NT * TB_DFF;               // 16,777,216
    const size_t DD  = (size_t)TB_D * TB_D;       // 1,048,576
    const size_t DF  = (size_t)TB_D * TB_DFF;     // 4,194,304

    // carve workspace
    char* ws = (char*)d_ws;
    size_t off = 0;
    auto abf = [&](size_t n) { __bf16* p = (__bf16*)(ws + off); off += ((n * 2 + 255) / 256) * 256; return p; };
    auto af32 = [&](size_t n) { float* p = (float*)(ws + off); off += ((n * 4 + 255) / 256) * 256; return p; };

    __bf16* xb   = abf(ND);
    __bf16* wqb  = abf(DD);
    __bf16* wkb  = abf(DD);
    __bf16* wvb  = abf(DD);
    __bf16* wob  = abf(DD);
    __bf16* w1b  = abf(DF);
    __bf16* w2b  = abf(DF);
    __bf16* Qb   = abf(ND);
    __bf16* Kb   = abf(ND);
    __bf16* Vb   = abf(ND);
    __bf16* ctxb = abf(ND);
    __bf16* hb   = abf(ND);
    __bf16* ffb  = abf(NF);
    float*  hpre = af32(ND);   // x + attn_out (pre-LN1); reused as y2 below
    float*  h    = af32(ND);   // post-LN1
    float*  y2   = hpre;       // h + ffn_out (pre-LN2) aliases hpre

    float* outF  = (float*)d_out;
    float* attnP = outF + ND;  // attention output region of the tuple

    // 1) fp32 -> bf16
    auto cvt = [&](const float* src, __bf16* dst, size_t n) {
        const size_t n4 = n / 4;
        cvt_f32_bf16<<<dim3((unsigned)((n4 + 255) / 256)), 256, 0, stream>>>(src, dst, (int)n4);
    };
    cvt(x, xb, ND);
    cvt(w_q, wqb, DD); cvt(w_k, wkb, DD); cvt(w_v, wvb, DD); cvt(w_o, wob, DD);
    cvt(w1, w1b, DF);  cvt(w2, w2b, DF);

    // 2) QKV projections, permuted to [B,H,S,dk]
    dim3 gQKV(NT / 128, TB_D / 128);
    gemm_bf16<0><<<gQKV, 256, 0, stream>>>(xb, wqb, (int)NT, TB_D, TB_D, nullptr, nullptr, nullptr, Qb);
    gemm_bf16<0><<<gQKV, 256, 0, stream>>>(xb, wkb, (int)NT, TB_D, TB_D, nullptr, nullptr, nullptr, Kb);
    gemm_bf16<0><<<gQKV, 256, 0, stream>>>(xb, wvb, (int)NT, TB_D, TB_D, nullptr, nullptr, nullptr, Vb);

    // 3) scores = QK^T / sqrt(dk), straight into d_out's attn region
    attn_scores<<<dim3(TB_S / 16, TB_S / 128, TB_B * TB_H), 256, 0, stream>>>(
        Qb, Kb, attnP, 0.125f);

    // 4) softmax rows in place
    softmax_rows<<<dim3((unsigned)(TB_B * TB_H * TB_S)), 256, 0, stream>>>(attnP);

    // 5) context = attn @ V, merged back to [B,S,D]
    attn_context<<<dim3(TB_S / 64, TB_B * TB_H), 256, 0, stream>>>(attnP, Vb, ctxb);

    // 6) out-proj + bias + residual(x) -> hpre
    dim3 gOP(NT / 128, TB_D / 128);
    gemm_bf16<1><<<gOP, 256, 0, stream>>>(ctxb, wob, (int)NT, TB_D, TB_D, b_o, x, hpre, nullptr);

    // 7) LN1 -> h (f32) + hb (bf16)
    layernorm_rows<<<dim3((unsigned)NT), 256, 0, stream>>>(hpre, ln1g, ln1b, h, hb);

    // 8) FFN1 + bias + exact GELU -> ffb (bf16)
    dim3 gF1(NT / 128, TB_DFF / 128);
    gemm_bf16<2><<<gF1, 256, 0, stream>>>(hb, w1b, (int)NT, TB_DFF, TB_D, b1, nullptr, nullptr, ffb);

    // 9) FFN2 + bias + residual(h) -> y2
    dim3 gF2(NT / 128, TB_D / 128);
    gemm_bf16<1><<<gF2, 256, 0, stream>>>(ffb, w2b, (int)NT, TB_D, TB_DFF, b2, h, y2, nullptr);

    // 10) LN2 -> final output
    layernorm_rows<<<dim3((unsigned)NT), 256, 0, stream>>>(y2, ln2g, ln2b, outF, nullptr);

    (void)in_sizes; (void)n_in; (void)out_size; (void)ws_size;
}
